// MultiHeadAttention_68539088110533
// MI455X (gfx1250) — matrix operand order repaired and checked by reading, not compiled, not run on hardware
//
#include <hip/hip_runtime.h>
#include <hip/hip_bf16.h>

// ---------------------------------------------------------------------------
// MHA forward for MI455X (gfx1250, wave32, WMMA 16x16x32 f16 -> f32 acc).
// B=2 S=2048 D=1024 H=16 HD=64.  Compute-bound (~70 GFLOP vs ~100MB HBM),
// so all matmuls go through v_wmma_f32_16x16x32_f16; weight tiles are staged
// block-wide into LDS with async global->LDS copies (ASYNCcnt) and
// double-buffered so the DMA overlaps the WMMAs.
// ---------------------------------------------------------------------------

#define B_  2
#define S_  2048
#define D_  1024
#define H_  16
#define HD_ 64
#define M_  (B_ * S_)   // 4096 rows for the projection GEMMs

typedef _Float16 v16h __attribute__((ext_vector_type(16)));
typedef _Float16 v8h  __attribute__((ext_vector_type(8)));
typedef _Float16 v4h  __attribute__((ext_vector_type(4)));
typedef float    v8f  __attribute__((ext_vector_type(8)));
typedef int      v4i  __attribute__((ext_vector_type(4)));

#if defined(__gfx1250__) && __has_builtin(__builtin_amdgcn_global_load_async_to_lds_b128)
#define USE_ASYNC_LDS 1
// clang prototype: (int4 addrspace(1)*, int4 addrspace(3)*, imm i32, imm i32)
typedef __attribute__((address_space(1))) v4i* gp4i;
typedef __attribute__((address_space(3))) v4i* lp4i;
// generic->as(1): same 64-bit value; generic->as(3): low 32 bits are the LDS
// byte offset, so integer casts are exact.
#define AS_GLOBAL(p) ((gp4i)(unsigned long long)(p))
#define AS_LDS(p)    ((lp4i)(unsigned int)(unsigned long long)(p))
#else
#define USE_ASYNC_LDS 0
#endif

static __device__ inline v8f wmma_f16(v16h a, v16h b, v8f c) {
  // D = A(16x32 f16) * B(32x16 f16) + C(16x16 f32)
  return __builtin_amdgcn_wmma_f32_16x16x32_f16(
      /*neg_a=*/false, a, /*neg_b=*/false, b,
      /*c_mod=*/(short)0, c, /*reuse_a=*/false, /*reuse_b=*/false);
}

// A-fragment (16x32, f16) from row-major [16][ldk]:
//  lane L: row M = L&15; elems 0..15 -> K = (L>=16?16:0)+0..15
static __device__ inline v16h load_a16x32(const _Float16* __restrict__ base,
                                          int ldk, int lane) {
  const int row = lane & 15;
  const int off = (lane >> 4) << 4;
  v8h lo = *(const v8h*)(base + (size_t)row * ldk + off);
  v8h hi = *(const v8h*)(base + (size_t)row * ldk + off + 8);
  v16h r;
#pragma unroll
  for (int i = 0; i < 8; ++i) { r[i] = lo[i]; r[i + 8] = hi[i]; }
  return r;
}

// B-fragment (32x16, f16): B[k,n] from row-major Wlike[n][ldk] at column k.
// lane L: n = L&15, k = 16*(L>>4) + e  -> one contiguous 32B load.
static __device__ inline v16h load_b32x16(const _Float16* __restrict__ base,
                                          int ldk, int lane) {
  const int col = lane & 15;
  const int off = (lane >> 4) << 4;
  return *(const v16h*)(base + (size_t)col * ldk + off);
}

// Same B-fragment but from an LDS tile [64 rows][40 halfs (padded)].
static __device__ inline v16h load_b_lds(const _Float16* tile, int lane, int t) {
  const int row = t * 16 + (lane & 15);
  const int off = (lane >> 4) << 4;
  v8h lo = *(const v8h*)(tile + row * 40 + off);
  v8h hi = *(const v8h*)(tile + row * 40 + off + 8);
  v16h r;
#pragma unroll
  for (int i = 0; i < 8; ++i) { r[i] = lo[i]; r[i + 8] = hi[i]; }
  return r;
}

// ---------------------------------------------------------------------------
// f32 -> f16 conversion (4 elems / thread)
// ---------------------------------------------------------------------------
__global__ void cvt_f32_f16(const float* __restrict__ src,
                            _Float16* __restrict__ dst, int n4) {
  int i = blockIdx.x * blockDim.x + threadIdx.x;
  if (i >= n4) return;
  float4 v = ((const float4*)src)[i];
  v4h h;
  h[0] = (_Float16)v.x; h[1] = (_Float16)v.y;
  h[2] = (_Float16)v.z; h[3] = (_Float16)v.w;
  *(v4h*)(dst + (size_t)i * 4) = h;
}

// ---------------------------------------------------------------------------
// C[m,n] = (sum_k X[m,k]*W[n,k] + bias[n]) * scale
// Block = 8 waves = 256 rows x 64 cols; wave = 32x64 tile (8 WMMAs/k-step).
// W tile (64x32 halfs) double-buffered in LDS, fetched block-wide with async
// global->LDS b128 copies issued one k-step ahead of consumption.
// mode 0: f16 out, head-split [B,H,S,HD]            (Q, K)
// mode 1: f16 out, head-split transposed [B,H,HD,S] (V)
// mode 2: f32 out, plain row-major [M,N]            (final projection)
// ---------------------------------------------------------------------------
__global__ void gemm_xwt(const _Float16* __restrict__ X,
                         const _Float16* __restrict__ W,
                         const float* __restrict__ bias,
                         void* __restrict__ out,
                         int Mrows, int Ncols, int K, int mode, float scale) {
  __shared__ alignas(16) _Float16 wtile[2][64 * 40];  // 2 x 5KB, padded rows
  const int lane = threadIdx.x & 31;
  const int wv   = threadIdx.x >> 5;
  const int nbn  = Ncols >> 6;
  const int bm   = blockIdx.x / nbn;
  const int bn   = blockIdx.x % nbn;
  const int n0   = bn << 6;
  const int m0   = (bm << 8) + (wv << 5);   // this wave's 32 rows

  // cooperative W-tile fetch: 256 threads x 16B = 64 rows x 64B
  const int trow   = threadIdx.x >> 2;          // 0..63
  const int tchunk = (threadIdx.x & 3) << 3;    // half-offset 0,8,16,24
  const _Float16* wsrc = W + (size_t)(n0 + trow) * K + tchunk;
  _Float16* wdst0 = &wtile[0][trow * 40 + tchunk];
  _Float16* wdst1 = &wtile[1][trow * 40 + tchunk];

#if USE_ASYNC_LDS
#define FETCH_W(buf, kk)                                                      \
  __builtin_amdgcn_global_load_async_to_lds_b128(                             \
      AS_GLOBAL(wsrc + (kk)), AS_LDS((buf) ? wdst1 : wdst0), 0, 0)
#define WAIT_W() __asm__ volatile("s_wait_asynccnt 0x0" ::: "memory")
#else
#define FETCH_W(buf, kk) *(v8h*)((buf) ? wdst1 : wdst0) = *(const v8h*)(wsrc + (kk))
#define WAIT_W()
#endif

  v8f acc[2][4] = {};
  FETCH_W(0, 0);
  const _Float16* xa0 = X + (size_t)m0 * K;
  const _Float16* xa1 = X + (size_t)(m0 + 16) * K;

  for (int k = 0, it = 0; k < K; k += 32, ++it) {
    const int cur = it & 1;
    WAIT_W();            // own async copies into `cur` complete
    __syncthreads();     // all waves' copies done; prev buffer reads done
    if (k + 32 < K) FETCH_W(cur ^ 1, k + 32);

    v16h a0 = load_a16x32(xa0 + k, K, lane);
    v16h a1 = load_a16x32(xa1 + k, K, lane);
    const _Float16* tile = &wtile[cur][0];
#pragma unroll
    for (int t = 0; t < 4; ++t) {
      v16h b = load_b_lds(tile, lane, t);
      acc[0][t] = wmma_f16(a0, b, acc[0][t]);
      acc[1][t] = wmma_f16(a1, b, acc[1][t]);
    }
    __syncthreads();     // reads of `cur` done before it is refilled
  }

  const int half = lane >> 4, n_lo = lane & 15;
#pragma unroll
  for (int ms = 0; ms < 2; ++ms) {
#pragma unroll
    for (int t = 0; t < 4; ++t) {
      const int n = n0 + t * 16 + n_lo;
      const float bv = bias[n];
#pragma unroll
      for (int r = 0; r < 8; ++r) {
        const int m = m0 + ms * 16 + r + (half << 3);  // D-layout row
        float val = (acc[ms][t][r] + bv) * scale;
        if (mode == 2) {
          ((float*)out)[(size_t)m * Ncols + n] = val;
        } else {
          const int bb = m >> 11;                  // m / S_
          const int s  = m & (S_ - 1);
          const int h  = n >> 6;                   // n / HD_
          const int d  = n & (HD_ - 1);
          _Float16 hv = (_Float16)val;
          if (mode == 0)
            ((_Float16*)out)[(((size_t)(bb * H_ + h)) * S_ + s) * HD_ + d] = hv;
          else
            ((_Float16*)out)[(((size_t)(bb * H_ + h)) * HD_ + d) * S_ + s] = hv;
        }
      }
    }
  }
  (void)Mrows;
}

// ---------------------------------------------------------------------------
// Flash attention: one wave per (b,h, 16-row q tile).  Causal, online
// softmax, k processed 32 columns per iteration (two 16x16 score tiles).
// Qh,Kh: [B,H,S,HD] f16 (Q pre-scaled by 1/sqrt(HD)); Vt: [B,H,HD,S] f16.
// Output: [B,S,H,HD] f16 (== [B,S,D] with head-major columns).
// ---------------------------------------------------------------------------
__global__ void flash_attn(const _Float16* __restrict__ Qh,
                           const _Float16* __restrict__ Kh,
                           const _Float16* __restrict__ Vt,
                           _Float16* __restrict__ O) {
  __shared__ alignas(16) _Float16 plds[4][16 * 40];  // per-wave P scratch
  const int lane = threadIdx.x & 31;
  const int wv   = threadIdx.x >> 5;
  const int wid  = blockIdx.x * 4 + wv;
  const int qt   = wid & 127;       // S_/16 = 128 q tiles
  const int bh   = wid >> 7;        // b*H + h
  const int q0   = qt << 4;
  const int half = lane >> 4, n_lo = lane & 15;

  const _Float16* qbase = Qh + ((size_t)bh * S_ + q0) * HD_;
  v16h qa0 = load_a16x32(qbase, HD_, lane);        // d = 0..31
  v16h qa1 = load_a16x32(qbase + 32, HD_, lane);   // d = 32..63

  v8f o[4] = {};
  float mi[8], li[8];
#pragma unroll
  for (int r = 0; r < 8; ++r) { mi[r] = -1e30f; li[r] = 0.f; }

  const _Float16* kh = Kh + (size_t)bh * S_ * HD_;
  const _Float16* vt = Vt + (size_t)bh * HD_ * S_;
  _Float16* my = plds[wv];

  const int kend = q0 + 16;                        // causal bound
  for (int kb = 0; kb < kend; kb += 32) {
    // ---- scores: S[q, kb..kb+31] = Q . K^T   (4 WMMAs) ----
    const _Float16* kt0 = kh + (size_t)kb * HD_;
    v8f s0 = {}, s1 = {};
    {
      v16h b00 = load_b32x16(kt0,                 HD_, lane);
      v16h b10 = load_b32x16(kt0 + 16 * HD_,      HD_, lane);
      s0 = wmma_f16(qa0, b00, s0);
      s1 = wmma_f16(qa0, b10, s1);
      v16h b01 = load_b32x16(kt0 + 32,            HD_, lane);
      v16h b11 = load_b32x16(kt0 + 16 * HD_ + 32, HD_, lane);
      s0 = wmma_f16(qa1, b01, s0);
      s1 = wmma_f16(qa1, b11, s1);
    }

    // ---- causal mask + online softmax (row stats replicated per 16-lane half)
    float p0[8], p1[8], rmax[8];
#pragma unroll
    for (int r = 0; r < 8; ++r) {
      const int qg = q0 + r + (half << 3);
      float x0 = s0[r]; if (kb + n_lo > qg)      x0 = -1e30f;
      float x1 = s1[r]; if (kb + 16 + n_lo > qg) x1 = -1e30f;
      p0[r] = x0; p1[r] = x1;
      rmax[r] = fmaxf(x0, x1);
    }
#pragma unroll
    for (int r = 0; r < 8; ++r)
#pragma unroll
      for (int sh = 1; sh < 16; sh <<= 1)
        rmax[r] = fmaxf(rmax[r], __shfl_xor(rmax[r], sh, 32));

    float alpha[8], rsum[8];
#pragma unroll
    for (int r = 0; r < 8; ++r) {
      const float mn = fmaxf(mi[r], rmax[r]);
      alpha[r] = __expf(mi[r] - mn);
      mi[r] = mn;
      const float e0 = __expf(p0[r] - mn);
      const float e1 = __expf(p1[r] - mn);
      p0[r] = e0; p1[r] = e1;
      rsum[r] = e0 + e1;
    }
#pragma unroll
    for (int r = 0; r < 8; ++r) {
#pragma unroll
      for (int sh = 1; sh < 16; sh <<= 1)
        rsum[r] += __shfl_xor(rsum[r], sh, 32);
      li[r] = li[r] * alpha[r] + rsum[r];
    }
#pragma unroll
    for (int t = 0; t < 4; ++t)
#pragma unroll
      for (int r = 0; r < 8; ++r) o[t][r] *= alpha[r];

    // ---- P: D-layout -> A-layout via per-wave LDS (wave-local, in-order DS)
#pragma unroll
    for (int r = 0; r < 8; ++r) {
      const int mrow = r + (half << 3);
      my[mrow * 40 + n_lo]      = (_Float16)p0[r];
      my[mrow * 40 + 16 + n_lo] = (_Float16)p1[r];
    }
    __asm__ volatile("s_wait_dscnt 0x0" ::: "memory");
    v16h pf;
    {
      const int row = lane & 15;
      const int off = half << 4;
      v8h lo = *(const v8h*)(my + row * 40 + off);
      v8h hi = *(const v8h*)(my + row * 40 + 8 + off);
#pragma unroll
      for (int i = 0; i < 8; ++i) { pf[i] = lo[i]; pf[i + 8] = hi[i]; }
    }

    // ---- O += P . V   (V stored [HD][S] so B-frags load contiguously)
#pragma unroll
    for (int t = 0; t < 4; ++t) {
      v16h vf = load_b32x16(vt + (size_t)(t * 16) * S_ + kb, S_, lane);
      o[t] = wmma_f16(pf, vf, o[t]);
    }
  }

  // ---- finalize and store [B,S,H,HD]
  const int bb = bh >> 4, h = bh & 15;
  float inv[8];
#pragma unroll
  for (int r = 0; r < 8; ++r) inv[r] = 1.f / li[r];
#pragma unroll
  for (int t = 0; t < 4; ++t) {
    const int d = t * 16 + n_lo;
#pragma unroll
    for (int r = 0; r < 8; ++r) {
      const int qg = q0 + r + (half << 3);
      O[(((size_t)bb * S_ + qg) * H_ + h) * HD_ + d] =
          (_Float16)(o[t][r] * inv[r]);
    }
  }
}

// ---------------------------------------------------------------------------
// Host launcher
// ---------------------------------------------------------------------------
extern "C" void kernel_launch(void* const* d_in, const int* in_sizes, int n_in,
                              void* d_out, int out_size, void* d_ws,
                              size_t ws_size, hipStream_t stream) {
  (void)in_sizes; (void)n_in; (void)out_size; (void)ws_size;

  // setup_inputs order: q k v mask Wq bq Wk bk Wv bv Wo bo
  const float* q  = (const float*)d_in[0];
  const float* k  = (const float*)d_in[1];
  const float* v  = (const float*)d_in[2];
  const float* Wq = (const float*)d_in[4];
  const float* bq = (const float*)d_in[5];
  const float* Wk = (const float*)d_in[6];
  const float* bk = (const float*)d_in[7];
  const float* Wv = (const float*)d_in[8];
  const float* bv = (const float*)d_in[9];
  const float* Wo = (const float*)d_in[10];
  const float* bo = (const float*)d_in[11];

  char* ws = (char*)d_ws;
  size_t off = 0;
  auto alloc = [&](size_t bytes) -> void* {
    void* p = ws + off;
    off += (bytes + 255) & ~(size_t)255;
    return p;
  };
  _Float16* q16  = (_Float16*)alloc((size_t)M_ * D_ * 2);
  _Float16* k16  = (_Float16*)alloc((size_t)M_ * D_ * 2);
  _Float16* v16  = (_Float16*)alloc((size_t)M_ * D_ * 2);
  _Float16* wq16 = (_Float16*)alloc((size_t)D_ * D_ * 2);
  _Float16* wk16 = (_Float16*)alloc((size_t)D_ * D_ * 2);
  _Float16* wv16 = (_Float16*)alloc((size_t)D_ * D_ * 2);
  _Float16* wo16 = (_Float16*)alloc((size_t)D_ * D_ * 2);
  _Float16* Qh   = (_Float16*)alloc((size_t)M_ * D_ * 2);  // [B,H,S,HD]
  _Float16* Kh   = (_Float16*)alloc((size_t)M_ * D_ * 2);  // [B,H,S,HD]
  _Float16* Vtr  = (_Float16*)alloc((size_t)M_ * D_ * 2);  // [B,H,HD,S]
  _Float16* att  = (_Float16*)alloc((size_t)M_ * D_ * 2);  // [B,S,D]

  const int T = 256;
  auto cvt = [&](const float* s, _Float16* dst, int n) {
    cvt_f32_f16<<<(n / 4 + T - 1) / T, T, 0, stream>>>(s, dst, n / 4);
  };
  cvt(q,  q16,  M_ * D_);
  cvt(k,  k16,  M_ * D_);
  cvt(v,  v16,  M_ * D_);
  cvt(Wq, wq16, D_ * D_);
  cvt(Wk, wk16, D_ * D_);
  cvt(Wv, wv16, D_ * D_);
  cvt(Wo, wo16, D_ * D_);

  // projections: (M/256) * (N/64) = 16*16 = 256 blocks of 8 waves
  const int gemm_blocks = (M_ / 256) * (D_ / 64);
  const float qscale = 0.125f;                        // 1/sqrt(HD)
  gemm_xwt<<<gemm_blocks, 256, 0, stream>>>(q16, wq16, bq, Qh,
                                            M_, D_, D_, 0, qscale);
  gemm_xwt<<<gemm_blocks, 256, 0, stream>>>(k16, wk16, bk, Kh,
                                            M_, D_, D_, 0, 1.0f);
  gemm_xwt<<<gemm_blocks, 256, 0, stream>>>(v16, wv16, bv, Vtr,
                                            M_, D_, D_, 1, 1.0f);

  // attention: B*H*(S/16) = 4096 waves, 4 waves per block
  flash_attn<<<1024, 128, 0, stream>>>(Qh, Kh, Vtr, att);

  // output projection -> fp32 [B,S,D]
  gemm_xwt<<<gemm_blocks, 256, 0, stream>>>(att, wo16, bo, d_out,
                                            M_, D_, D_, 2, 1.0f);
}
